// TambaMambaEncoder_36412732736030
// MI455X (gfx1250) — matrix operand order repaired and checked by reading, not compiled, not run on hardware
//
#include <hip/hip_runtime.h>
#include <hip/hip_bf16.h>
#include <math.h>

// ---------------- constants ----------------
#define BB      8
#define LL      8192
#define BL      (BB * LL)        // 65536 tokens
#define DM      128
#define DI      256
#define DS      16
#define DCONV   4
#define DTR     8
#define NPROJ   (2 * DI)         // 512
#define NCOMB   (DS + DS + DI)   // 288 : [Bc(16) | Cc(16) | dt_lin(256)]

typedef __attribute__((ext_vector_type(16))) _Float16 v16h;
typedef __attribute__((ext_vector_type(8)))  _Float16 v8h;
typedef __attribute__((ext_vector_type(8)))  float    v8f;

union Frag16 { v16h v; v8h h[2]; };

// ---------------- prep kernels ----------------
__global__ void k_f32_to_f16(const float* __restrict__ src, _Float16* __restrict__ dst, int n) {
    for (int i = blockIdx.x * blockDim.x + threadIdx.x; i < n; i += gridDim.x * blockDim.x)
        dst[i] = (_Float16)src[i];
}

// Build combined x-proj weight Wc (per layer, 288 x 256, f16):
//   rows 0..31  : x_proj_w rows 8..39        (Bc then Cc)
//   rows 32..287: M[d,e] = sum_r dtw[d,r]*xw[r,e]   (folds dt_proj into x_proj)
__global__ void k_build_wc(const float* __restrict__ xw, const float* __restrict__ dtw,
                           _Float16* __restrict__ wc) {
    int tid = blockIdx.x * blockDim.x + threadIdx.x;
    int total = 2 * NCOMB * DI;
    if (tid >= total) return;
    int l = tid / (NCOMB * DI);
    int rem = tid - l * (NCOMB * DI);
    int n = rem / DI;
    int e = rem - n * DI;
    const float* xwl  = xw  + (size_t)l * (DTR + 2 * DS) * DI;
    const float* dtwl = dtw + (size_t)l * DI * DTR;
    float v;
    if (n < 32) {
        v = xwl[(size_t)(DTR + n) * DI + e];
    } else {
        int d = n - 32;
        float s = 0.f;
        #pragma unroll
        for (int r = 0; r < DTR; ++r) s += dtwl[d * DTR + r] * xwl[(size_t)r * DI + e];
        v = s;
    }
    wc[(size_t)l * NCOMB * DI + (size_t)n * DI + e] = (_Float16)v;
}

__global__ void k_build_A(const float* __restrict__ a_log, float* __restrict__ a_neg, int n) {
    for (int i = blockIdx.x * blockDim.x + threadIdx.x; i < n; i += gridDim.x * blockDim.x)
        a_neg[i] = -__expf(a_log[i]);
}

// ---------------- RMSNorm (one wave32 per token, d=128 -> float4/lane) ----------------
template <bool F16OUT>
__global__ __launch_bounds__(256) void k_rmsnorm(const float* __restrict__ h,
                                                 const float* __restrict__ w,
                                                 void* __restrict__ out) {
    int wave = threadIdx.x >> 5, lane = threadIdx.x & 31;
    int tok = blockIdx.x * 8 + wave;
    const float4 hv = *(const float4*)(h + (size_t)tok * DM + lane * 4);
    float ss = hv.x * hv.x + hv.y * hv.y + hv.z * hv.z + hv.w * hv.w;
    #pragma unroll
    for (int o = 16; o >= 1; o >>= 1) ss += __shfl_xor(ss, o, 32);
    float r = rsqrtf(ss * (1.f / DM) + 1e-5f);
    const float4 wv = *(const float4*)(w + lane * 4);
    if (F16OUT) {
        _Float16* o16 = (_Float16*)out + (size_t)tok * DM + lane * 4;
        o16[0] = (_Float16)(hv.x * r * wv.x);
        o16[1] = (_Float16)(hv.y * r * wv.y);
        o16[2] = (_Float16)(hv.z * r * wv.z);
        o16[3] = (_Float16)(hv.w * r * wv.w);
    } else {
        float4 ov = { hv.x * r * wv.x, hv.y * r * wv.y, hv.z * r * wv.z, hv.w * r * wv.w };
        *((float4*)out + ((size_t)tok * DM + lane * 4) / 4) = ov;
    }
}

// ---------------- WMMA GEMM: C(M,N) = A(M,K) * W(N,K)^T ----------------
// A: f16 row-major (K contig). W: f16 row-major (K contig). 8 waves/block,
// 16 tokens/wave, A fragments register-cached across all N tiles.
// MODE 0: plain store to out0 (ldc=512)           [in_proj]
// MODE 1: split store Bc/Cc + softplus(dt+bias)   [combined x_proj]
// MODE 2: out0 = resid + acc (ldc=128)            [out_proj + residual]
template <int KCHUNKS, int MODE>
__global__ __launch_bounds__(256) void k_gemm(const _Float16* __restrict__ A,
                                              const _Float16* __restrict__ W,
                                              int nTiles,
                                              float* __restrict__ out0,
                                              float* __restrict__ outB,
                                              float* __restrict__ outC,
                                              float* __restrict__ outDt,
                                              const float* __restrict__ bias,
                                              const float* __restrict__ resid) {
    constexpr int K = KCHUNKS * 32;
    const int lane = threadIdx.x & 31;
    const int wave = threadIdx.x >> 5;
    const int half = lane >> 4;     // 0: lanes 0-15, 1: lanes 16-31
    const int r    = lane & 15;
    const int m0   = blockIdx.x * 128 + wave * 16;

    // A fragment (ISA 16-bit 16x32 layout): lane holds row M=r;
    // halves 0..7 -> K = k0 + 8*half + j ; halves 8..15 -> K = k0 + 16 + 8*half + j
    Frag16 af[KCHUNKS];
    const _Float16* arow = A + (size_t)(m0 + r) * K + half * 8;
    #pragma unroll
    for (int kc = 0; kc < KCHUNKS; ++kc) {
        af[kc].h[0] = *(const v8h*)(arow + kc * 32);
        af[kc].h[1] = *(const v8h*)(arow + kc * 32 + 16);
    }

    for (int nt = 0; nt < nTiles; ++nt) {
        const int n0 = nt * 16;
        v8f acc = {};
        // B fragment: lane holds col N=n0+r; halves 0..7 -> K = k0 + 8*half + j ; halves 8..15 -> K = k0 + 16 + 8*half + j
        const _Float16* wrow = W + (size_t)(n0 + r) * K + half * 8;
        #pragma unroll
        for (int kc = 0; kc < KCHUNKS; ++kc) {
            Frag16 bf;
            bf.h[0] = *(const v8h*)(wrow + kc * 32);
            bf.h[1] = *(const v8h*)(wrow + kc * 32 + 16);
            acc = __builtin_amdgcn_wmma_f32_16x16x32_f16(false, af[kc].v, false, bf.v,
                                                         (short)0, acc, false, false);
        }
        // Epilogue. D layout: lane -> N = n0+r; VGPR vr -> M = m0 + vr + 8*half.
        const int col = n0 + r;
        #pragma unroll
        for (int vr = 0; vr < 8; ++vr) {
            const int tok = m0 + vr + 8 * half;
            const float v = acc[vr];
            if (MODE == 0) {
                out0[(size_t)tok * NPROJ + col] = v;
            } else if (MODE == 1) {
                if (col < DS) {
                    outB[(size_t)tok * DS + col] = v;
                } else if (col < 2 * DS) {
                    outC[(size_t)tok * DS + (col - DS)] = v;
                } else {
                    const int d = col - 2 * DS;
                    const float t = v + bias[d];
                    outDt[(size_t)tok * DI + d] = (t > 20.f) ? t : logf(1.f + __expf(t));
                }
            } else {
                const size_t i = (size_t)tok * DM + col;
                out0[i] = resid[i] + v;
            }
        }
    }
}

// ---------------- causal depthwise conv (taps=4) + bias + SiLU ----------------
// blockDim.x = 256 (= d), each thread produces 8 consecutive l for one (b,d).
__global__ __launch_bounds__(256) void k_conv(const float* __restrict__ proj,
                                              const float* __restrict__ cw,
                                              const float* __restrict__ cb,
                                              float* __restrict__ xh32,
                                              _Float16* __restrict__ xh16) {
    const int d = threadIdx.x;
    const int chunk = blockIdx.x;
    const int b = chunk / (LL / 8);
    const int l0 = (chunk - b * (LL / 8)) * 8;
    const float w0 = cw[d * DCONV + 0], w1 = cw[d * DCONV + 1];
    const float w2 = cw[d * DCONV + 2], w3 = cw[d * DCONV + 3];
    const float bias = cb[d];
    float v[11];
    #pragma unroll
    for (int i = 0; i < 11; ++i) {
        const int l = l0 - 3 + i;
        v[i] = (l >= 0) ? proj[((size_t)(b * LL + l)) * NPROJ + d] : 0.f;
    }
    #pragma unroll
    for (int j = 0; j < 8; ++j) {
        const float s = v[j] * w0 + v[j + 1] * w1 + v[j + 2] * w2 + v[j + 3] * w3 + bias;
        const float act = s / (1.f + __expf(-s));  // silu
        const size_t idx = ((size_t)(b * LL + l0 + j)) * DI + d;
        xh32[idx] = act;
        xh16[idx] = (_Float16)act;
    }
}

// ---------------- selective scan, fused skip + SiLU gate ----------------
// One 16-lane half-wave per (b,d) channel; lane = state index s.
// h_s <- exp(dt*A_s)*h_s + dt*x*B_s ; y = sum_s h_s*C_s (4x shfl_xor reduce).
__global__ __launch_bounds__(256) void k_scan(const float* __restrict__ dt,
                                              const float* __restrict__ Bc,
                                              const float* __restrict__ Cc,
                                              const float* __restrict__ xh32,
                                              const float* __restrict__ proj,
                                              const float* __restrict__ a_neg,
                                              const float* __restrict__ dsk,
                                              _Float16* __restrict__ yg16) {
    const int s  = threadIdx.x & 15;
    const int ch = blockIdx.x * 16 + (threadIdx.x >> 4);
    const int b = ch >> 8;
    const int d = ch & 255;
    const float a = a_neg[d * DS + s];
    const float dskip = dsk[d];
    float h = 0.f;
    const size_t base = (size_t)b * LL;
    #pragma unroll 2
    for (int l = 0; l < LL; ++l) {
        const size_t t = base + l;
        const float dtv = dt[t * DI + d];
        const float xv  = xh32[t * DI + d];
        const float bv  = Bc[t * DS + s];
        const float cv  = Cc[t * DS + s];
        h = __expf(dtv * a) * h + (dtv * xv) * bv;
        float y = h * cv;
        y += __shfl_xor(y, 8, 32);
        y += __shfl_xor(y, 4, 32);
        y += __shfl_xor(y, 2, 32);
        y += __shfl_xor(y, 1, 32);
        if (s == 0) {
            y += xv * dskip;
            const float z = proj[t * NPROJ + DI + d];
            const float g = z / (1.f + __expf(-z));      // silu(z)
            yg16[t * DI + d] = (_Float16)(y * g);
        }
    }
}

// ---------------- host launcher ----------------
extern "C" void kernel_launch(void* const* d_in, const int* in_sizes, int n_in,
                              void* d_out, int out_size, void* d_ws, size_t ws_size,
                              hipStream_t stream) {
    const float* x      = (const float*)d_in[0];
    const float* norm_w = (const float*)d_in[1];
    const float* in_w   = (const float*)d_in[2];
    const float* conv_w = (const float*)d_in[3];
    const float* conv_b = (const float*)d_in[4];
    const float* xp_w   = (const float*)d_in[5];
    const float* dt_w   = (const float*)d_in[6];
    const float* dt_b   = (const float*)d_in[7];
    const float* a_log  = (const float*)d_in[8];
    const float* d_skip = (const float*)d_in[9];
    const float* out_w  = (const float*)d_in[10];
    const float* normf  = (const float*)d_in[11];

    // workspace carve
    char* ws = (char*)d_ws;
    size_t off = 0;
    auto carve = [&](size_t bytes) -> char* {
        char* p = ws + off;
        off = (off + bytes + 255) & ~(size_t)255;
        return p;
    };
    float*    h     = (float*)   carve((size_t)BL * DM * 4);
    float*    proj  = (float*)   carve((size_t)BL * NPROJ * 4);
    float*    xh32  = (float*)   carve((size_t)BL * DI * 4);
    float*    dt    = (float*)   carve((size_t)BL * DI * 4);
    float*    Bcb   = (float*)   carve((size_t)BL * DS * 4);
    float*    Ccb   = (float*)   carve((size_t)BL * DS * 4);
    _Float16* hn16  = (_Float16*)carve((size_t)BL * DM * 2);
    _Float16* xh16  = (_Float16*)carve((size_t)BL * DI * 2);
    _Float16* yg16  = (_Float16*)carve((size_t)BL * DI * 2);
    _Float16* in16  = (_Float16*)carve((size_t)2 * NPROJ * DM * 2);
    _Float16* wc16  = (_Float16*)carve((size_t)2 * NCOMB * DI * 2);
    _Float16* ow16  = (_Float16*)carve((size_t)2 * DM * DI * 2);
    float*    aneg  = (float*)   carve((size_t)2 * DI * DS * 4);

    // ---- one-time weight prep (cheap; runs every call for determinism) ----
    {
        int n1 = 2 * NPROJ * DM;
        k_f32_to_f16<<<(n1 + 255) / 256, 256, 0, stream>>>(in_w, in16, n1);
        int n2 = 2 * DM * DI;
        k_f32_to_f16<<<(n2 + 255) / 256, 256, 0, stream>>>(out_w, ow16, n2);
        int n3 = 2 * NCOMB * DI;
        k_build_wc<<<(n3 + 255) / 256, 256, 0, stream>>>(xp_w, dt_w, wc16);
        int n4 = 2 * DI * DS;
        k_build_A<<<(n4 + 255) / 256, 256, 0, stream>>>(a_log, aneg, n4);
    }
    hipMemcpyAsync(h, x, (size_t)BL * DM * 4, hipMemcpyDeviceToDevice, stream);

    const dim3 blk(256);
    const dim3 grid_norm(BL / 8);
    const dim3 grid_gemm(BL / 128);
    const dim3 grid_conv(BB * (LL / 8));
    const dim3 grid_scan((BB * DI) / 16);

    for (int l = 0; l < 2; ++l) {
        // 1) RMSNorm -> f16
        k_rmsnorm<true><<<grid_norm, blk, 0, stream>>>(h, norm_w + (size_t)l * DM, hn16);
        // 2) in_proj: (BL,128) x (512,128)^T -> proj (BL,512)   [K=128 -> 4 chunks]
        k_gemm<4, 0><<<grid_gemm, blk, 0, stream>>>(hn16, in16 + (size_t)l * NPROJ * DM,
                                                    NPROJ / 16, proj,
                                                    nullptr, nullptr, nullptr, nullptr, nullptr);
        // 3) causal conv + SiLU -> xh32 / xh16
        k_conv<<<grid_conv, blk, 0, stream>>>(proj, conv_w + (size_t)l * DI * DCONV,
                                              conv_b + (size_t)l * DI, xh32, xh16);
        // 4) combined x_proj (+folded dt_proj): (BL,256) x (288,256)^T  [K=256 -> 8 chunks]
        k_gemm<8, 1><<<grid_gemm, blk, 0, stream>>>(xh16, wc16 + (size_t)l * NCOMB * DI,
                                                    NCOMB / 16, nullptr,
                                                    Bcb, Ccb, dt, dt_b + (size_t)l * DI, nullptr);
        // 5) selective scan, fused D-skip + silu(z) gate -> yg16
        k_scan<<<grid_scan, blk, 0, stream>>>(dt, Bcb, Ccb, xh32, proj,
                                              aneg + (size_t)l * DI * DS,
                                              d_skip + (size_t)l * DI, yg16);
        // 6) out_proj + residual: h += yg @ (128,256)^T   [K=256 -> 8 chunks]
        k_gemm<8, 2><<<grid_gemm, blk, 0, stream>>>(yg16, ow16 + (size_t)l * DM * DI,
                                                    DM / 16, h,
                                                    nullptr, nullptr, nullptr, nullptr, h);
    }
    // final RMSNorm -> d_out (fp32)
    k_rmsnorm<false><<<grid_norm, blk, 0, stream>>>(h, normf, d_out);
}
